// GlimpseNetwork_43284680409701
// MI455X (gfx1250) — matrix operand order repaired and checked by reading, not compiled, not run on hardware
//
#include <hip/hip_runtime.h>
#include <hip/hip_bf16.h>

// ---------------------------------------------------------------------------
// GlimpseNetwork for MI455X (gfx1250, wave32, WMMA)
//   K0: W_h (1024x704 f32) -> bf16 once (read-once weight, 1.4 MB)
//   K1: glimpse extraction + separable bicubic (LDS-staged, chunked) +
//       loc embedding -> feat (128x704) bf16
//   K2: out = feat @ W_h^T + b_h via v_wmma_f32_16x16x32_bf16,
//       32x32 macro-tile per wave (2x2 register blocking, packed b128 loads)
// ---------------------------------------------------------------------------

#define IMG        224
#define CH_STRIDE  (IMG * IMG)
#define G0         12
#define FLAT       576          // 4 * 12 * 12
#define IN_DIM     704          // = 22 * 32  (WMMA-K friendly)
#define H_SIZE     1024
#define BATCH      128
#define CW         72           // LDS staging chunk width (135x73 f32 = 39 KB)

typedef __attribute__((ext_vector_type(16))) __bf16 v16bf;
typedef __attribute__((ext_vector_type(8)))  __bf16 v8bf;
typedef __attribute__((ext_vector_type(8)))  float  v8f;

static __device__ __forceinline__ __bf16 f2bf(float f) {
    // round-to-nearest-even fp32 -> bf16 (portable bit trick)
    union { float f; unsigned u; } c; c.f = f;
    unsigned r = (c.u + 0x7FFFu + ((c.u >> 16) & 1u)) >> 16;
    unsigned short hs = (unsigned short)r;
    __bf16 h;
    __builtin_memcpy(&h, &hs, 2);
    return h;
}

// ---------------------------------------------------------------------------
// K0: one-time weight conversion, coalesced, 4 elements per thread.
// ---------------------------------------------------------------------------
__global__ void __launch_bounds__(256)
wconv_kernel(const float* __restrict__ Wh, __bf16* __restrict__ Wbf)
{
    const int N = H_SIZE * IN_DIM;                 // 720896
    const int stride = 704 * 256;                  // grid 704 x 256 -> 4 iters
    for (int i = blockIdx.x * 256 + threadIdx.x; i < N; i += stride)
        Wbf[i] = f2bf(Wh[i]);
}

// ---------------------------------------------------------------------------
// K1: one block per batch element.
// Separable bicubic: Out[i][j] = scale * sum_u k[u] * sum_v k[v] * S[i+u][j+v]
// with S = channel-summed crop, scale = 1/(3*(sum k)^2).
// S is staged into LDS in <=CW-wide column chunks (bounds/zero-pad applied
// once per element); column pass U and row pass then run entirely from LDS.
// ---------------------------------------------------------------------------
__global__ void __launch_bounds__(256)
glimpse_feat_kernel(const float* __restrict__ imgs,
                    const int*   __restrict__ locs,
                    const float* __restrict__ x_table,
                    const float* __restrict__ y_table,
                    __bf16*      __restrict__ feat)
{
    __shared__ float kv[124];           // taps (max ksz = 124)
    __shared__ float S[135 * (CW + 1)]; // staged channel-summed crop chunk
    __shared__ float U[135 * G0];       // column-pass buffer
    __shared__ float s_scale;

    const int b    = blockIdx.x;
    const int tid  = threadIdx.x;
    const int loc0 = locs[2 * b + 0];   // row (H)
    const int loc1 = locs[2 * b + 1];   // col (W)

    const float* img  = imgs + (size_t)b * 3 * CH_STRIDE;
    __bf16*      frow = feat + (size_t)b * IN_DIM;

    // location embedding: [x_table[loc0] | y_table[loc1]]
    for (int t = tid; t < 128; t += blockDim.x) {
        float v = (t < 64) ? x_table[loc0 * 64 + t]
                           : y_table[loc1 * 64 + (t - 64)];
        frow[FLAT + t] = f2bf(v);
    }

    const int GS[4] = {12, 23, 68, 135};
    for (int p = 0; p < 4; ++p) {
        const int g   = GS[p];
        const int ksz = g - G0 + 1;     // 1, 12, 57, 124

        // bicubic taps in fp64 (matches numpy constants)
        if (tid < ksz) {
            const double Ac = -0.5;
            double x = fabs(((double)tid - (double)ksz * 0.5 + 0.5) * 4.0 / (double)ksz);
            double k;
            if (x > 2.0)      k = 0.0;
            else if (x > 1.0) k = Ac*x*x*x - 5.0*Ac*x*x + 8.0*Ac*x - 4.0*Ac;
            else              k = (Ac + 2.0)*x*x*x - (Ac + 3.0)*x*x + 1.0;
            kv[tid] = (float)k;
        }
        __syncthreads();
        if (tid == 0) {
            double s = 0.0;
            for (int i = 0; i < ksz; ++i) s += (double)kv[i];
            s_scale = (float)(1.0 / (3.0 * s * s));  // folds 1/IN_CH and K.sum()
        }

        // zero the column-pass accumulator
        for (int idx = tid; idx < g * G0; idx += blockDim.x) U[idx] = 0.f;
        __syncthreads();

        const int r0 = loc0 - g / 2;    // crop origin (may be out of image)
        const int c0 = loc1 - g / 2;

        // chunked over crop columns: stage S, then accumulate U from LDS
        for (int x0 = 0; x0 < g; x0 += CW) {
            const int cw = (g - x0 < CW) ? (g - x0) : CW;

            for (int idx = tid; idx < g * cw; idx += blockDim.x) {
                const int y = idx / cw, x = idx % cw;
                const int row = r0 + y, col = c0 + x0 + x;
                float v = 0.f;
                if (row >= 0 && row < IMG && col >= 0 && col < IMG) {
                    const float* pr = img + row * IMG + col;
                    v = pr[0] + pr[CH_STRIDE] + pr[2 * CH_STRIDE];
                }
                S[y * (CW + 1) + x] = v;     // zero outside (matches jnp.pad)
            }
            __syncthreads();

            for (int idx = tid; idx < g * G0; idx += blockDim.x) {
                const int y = idx / G0, j = idx % G0;
                int vlo = x0 - j;          if (vlo < 0)   vlo = 0;
                int vhi = x0 + cw - j;     if (vhi > ksz) vhi = ksz;
                float acc = 0.f;
                const float* srow = &S[y * (CW + 1) - x0 + j];
                for (int v = vlo; v < vhi; ++v) acc += kv[v] * srow[v];
                U[idx] += acc;
            }
            __syncthreads();
        }

        // row pass: Out[i][j] = scale * sum_u kv[u] * U[i+u][j]
        for (int idx = tid; idx < G0 * G0; idx += blockDim.x) {
            const int i = idx / G0, j = idx % G0;
            float acc = 0.f;
            for (int u = 0; u < ksz; ++u) acc += kv[u] * U[(i + u) * G0 + j];
            frow[p * (G0 * G0) + idx] = f2bf(acc * s_scale);
        }
        __syncthreads();
    }
}

// ---------------------------------------------------------------------------
// K2: out(128,1024) = feat(128,704) @ W^T(704,1024) + b_h.
// One wave32 per 32x32 macro-tile (2x2 x 16x16 WMMA register blocking):
//   - A/B fragments are K-contiguous per lane -> pure global_load_b128
//   - 4 independent accumulation chains pipeline the WMMA unit
// Fragment layouts per CDNA5 ISA 7.12.2 (bf16):
//   A (16x32):  M = lane&15; V0..3 = K (hi?8:0)+0..7, V4..7 = K (hi?24:16)+0..7
//   B (32x16):  N = lane&15; same K order as A; this code loads A and B as K (hi?16:0)+0..15
//   C/D (f32):  Vr -> M = r + (hi?8:0), N = lane&15
// ---------------------------------------------------------------------------
static __device__ __forceinline__ v16bf cat8(v8bf lo, v8bf hi) {
    return __builtin_shufflevector(lo, hi, 0,1,2,3,4,5,6,7,8,9,10,11,12,13,14,15);
}

__global__ void __launch_bounds__(256)
gemm_wmma_kernel(const __bf16* __restrict__ feat,  // (128,704) bf16
                 const __bf16* __restrict__ Wbf,   // (1024,704) bf16
                 const float*  __restrict__ bh,    // (1024) f32
                 float*        __restrict__ out)   // (128,1024) f32
{
    const int lane = threadIdx.x & 31;
    const int wave = threadIdx.x >> 5;
    const int w    = blockIdx.x * 8 + wave;   // 16 blocks * 8 waves = 128 tiles
    const int mt2  = w >> 5;                  // 0..3   (32-row macro tiles)
    const int nt2  = w & 31;                  // 0..31  (32-col macro tiles)
    const int hi   = lane >> 4;
    const int l15  = lane & 15;

    const int m0 = mt2 * 32;
    const int n0 = nt2 * 32;

    const __bf16* arow0 = feat + (size_t)(m0 + l15) * IN_DIM;
    const __bf16* arow1 = arow0 + 16 * IN_DIM;
    const __bf16* brow0 = Wbf  + (size_t)(n0 + l15) * IN_DIM;  // B[k][n]=W[n][k]
    const __bf16* brow1 = brow0 + 16 * IN_DIM;

    v8f acc00 = {}, acc01 = {}, acc10 = {}, acc11 = {};

    for (int kt = 0; kt < IN_DIM / 32; ++kt) {
        const int ka0 = kt * 32 + (hi ? 16 : 0);    // 16B aligned
        const int ka1 = ka0 + 8;
        const int kb  = kt * 32 + (hi ? 16 : 0);

        v16bf a0 = cat8(*(const v8bf*)(arow0 + ka0), *(const v8bf*)(arow0 + ka1));
        v16bf a1 = cat8(*(const v8bf*)(arow1 + ka0), *(const v8bf*)(arow1 + ka1));
        v16bf b0 = cat8(*(const v8bf*)(brow0 + kb ), *(const v8bf*)(brow0 + kb + 8));
        v16bf b1 = cat8(*(const v8bf*)(brow1 + kb ), *(const v8bf*)(brow1 + kb + 8));

        // 8 args: (neg_a, A, neg_b, B, c_mod, C, reuse_a, reuse_b)
        acc00 = __builtin_amdgcn_wmma_f32_16x16x32_bf16(false, a0, false, b0,
                                                        (short)0, acc00, false, false);
        acc01 = __builtin_amdgcn_wmma_f32_16x16x32_bf16(false, a0, false, b1,
                                                        (short)0, acc01, false, false);
        acc10 = __builtin_amdgcn_wmma_f32_16x16x32_bf16(false, a1, false, b0,
                                                        (short)0, acc10, false, false);
        acc11 = __builtin_amdgcn_wmma_f32_16x16x32_bf16(false, a1, false, b1,
                                                        (short)0, acc11, false, false);
    }

    const float bias0 = bh[n0 + l15];
    const float bias1 = bh[n0 + 16 + l15];
#pragma unroll
    for (int r = 0; r < 8; ++r) {
        const int mA = m0 + r + (hi ? 8 : 0);
        const int mB = mA + 16;
        out[(size_t)mA * H_SIZE + n0 + l15]      = acc00[r] + bias0;
        out[(size_t)mA * H_SIZE + n0 + 16 + l15] = acc01[r] + bias1;
        out[(size_t)mB * H_SIZE + n0 + l15]      = acc10[r] + bias0;
        out[(size_t)mB * H_SIZE + n0 + 16 + l15] = acc11[r] + bias1;
    }
}

// ---------------------------------------------------------------------------
extern "C" void kernel_launch(void* const* d_in, const int* in_sizes, int n_in,
                              void* d_out, int out_size, void* d_ws, size_t ws_size,
                              hipStream_t stream)
{
    const float* imgs = (const float*)d_in[0];   // (128,3,224,224) f32
    const int*   locs = (const int*)  d_in[1];   // (128,2) i32
    const float* Wh   = (const float*)d_in[2];   // (1024,704) f32
    const float* bh   = (const float*)d_in[3];   // (1024)     f32
    const float* xt   = (const float*)d_in[4];   // (224,64)   f32
    const float* yt   = (const float*)d_in[5];   // (224,64)   f32

    // workspace layout (rewritten every call):
    //   [0, 176KB)        feat  (128 x 704) bf16
    //   [184320, +1.41MB) W_h   (1024 x 704) bf16
    __bf16* featbf = (__bf16*)d_ws;
    __bf16* wbf    = (__bf16*)((char*)d_ws + 184320);
    float*  outp   = (float*)d_out;

    wconv_kernel       <<<704,   256, 0, stream>>>(Wh, wbf);
    glimpse_feat_kernel<<<BATCH, 256, 0, stream>>>(imgs, locs, xt, yt, featbf);
    gemm_wmma_kernel   <<<16,    256, 0, stream>>>(featbf, wbf, bh, outp);
}
